// Speller_57939108823193
// MI455X (gfx1250) — matrix operand order repaired and checked by reading, not compiled, not run on hardware
//
#include <hip/hip_runtime.h>

#define TDEC  128
#define TENC  512
#define BATCH 32
#define HID   1024
#define EMBD  512
#define VOC   5000
#define VPAD  5120          // 320 tiles of 16 -> multiple of 8 tiles, no tail guards
#define KCOMB 2560          // E + H(ctx) + H(h)
#define NGATE 4096
#define KCLS  2048

typedef __attribute__((ext_vector_type(16))) __bf16 v16bf;
typedef __attribute__((ext_vector_type(8)))  __bf16 v8bf;
typedef __attribute__((ext_vector_type(8)))  float  v8f;

__device__ __forceinline__ __bf16 f2bf(float f) {
  union { float f; unsigned u; } in; in.f = f;
  unsigned u = in.u;
  unsigned r = (u + 0x7FFFu + ((u >> 16) & 1u)) >> 16;
  union { unsigned short s; __bf16 b; } out; out.s = (unsigned short)r;
  return out.b;
}

// ---------------- weight conversion (once per launch) ----------------

__global__ void __launch_bounds__(256)
conv_wcomb_kernel(const float* __restrict__ W_ih, const float* __restrict__ W_hh,
                  __bf16* __restrict__ out) {
  size_t i = (size_t)blockIdx.x * 256 + threadIdx.x;
  if (i >= (size_t)NGATE * KCOMB) return;
  int n = (int)(i / KCOMB), k = (int)(i % KCOMB);
  float v = (k < EMBD + HID) ? W_ih[(size_t)n * (EMBD + HID) + k]
                             : W_hh[(size_t)n * HID + (k - EMBD - HID)];
  out[i] = f2bf(v);
}

__global__ void __launch_bounds__(256)
conv_wcls_kernel(const float* __restrict__ W, __bf16* __restrict__ out) {
  size_t i = (size_t)blockIdx.x * 256 + threadIdx.x;
  if (i >= (size_t)VPAD * KCLS) return;
  int n = (int)(i / KCLS), k = (int)(i % KCLS);
  out[i] = f2bf(n < VOC ? W[(size_t)n * KCLS + k] : 0.f);
}

__global__ void __launch_bounds__(256)
zero_kernel(float* __restrict__ p, int n) {
  int i = blockIdx.x * 256 + threadIdx.x;
  if (i < n) p[i] = 0.f;
}

// -------- WMMA GEMM: Cpart[kh][32,N] = A[32, kh-half of K] * W[N,K]^T --------
// 4 waves/block: wave -> (m_tile = wave&1, two consecutive n-tiles).
// blockIdx.y = K-half; partial sums go to separate slabs (summed by consumer).
// No guards: grid exactly covers Ntiles, so EXEC is all-ones at every WMMA.

__global__ void __launch_bounds__(128)
wmma_gemm_bf16(const __bf16* __restrict__ A, const __bf16* __restrict__ W,
               float* __restrict__ C, int K, int ldc) {
  const int wave = threadIdx.x >> 5;
  const int lane = threadIdx.x & 31;
  const int m_tile = wave & 1;
  const int n0 = blockIdx.x * 4 + (wave >> 1) * 2;
  const int g = lane >> 4;
  const int r = lane & 15;
  const int kh = blockIdx.y;
  const int Khalf = K >> 1;

  const __bf16* arow = A + (size_t)(m_tile * 16 + r) * K + kh * Khalf + g * 16;
  const __bf16* w0   = W + (size_t)(n0 * 16 + r) * K + kh * Khalf + g * 16;
  const __bf16* w1   = w0 + (size_t)16 * K;

  v8f acc0 = {}; v8f acc1 = {};
  // software pipeline: fragment registers loaded one iteration ahead
  v8bf  a0 = *(const v8bf*)(arow);          // K = g*16 + [0..7]
  v8bf  a1 = *(const v8bf*)(arow + 8);      // K = g*16 + 8 + [0..7]
  v16bf b0 = *(const v16bf*)(w0);           // N = n0*16+r,    K = g*16 + [0..15]
  v16bf b1 = *(const v16bf*)(w1);           // N = n0*16+16+r
  for (int kt = 32; kt < Khalf; kt += 32) {
    v16bf a = __builtin_shufflevector(a0, a1, 0,1,2,3,4,5,6,7,8,9,10,11,12,13,14,15);
    v16bf c0 = b0, c1 = b1;
    a0 = *(const v8bf*)(arow + kt);
    a1 = *(const v8bf*)(arow + kt + 8);
    b0 = *(const v16bf*)(w0 + kt);
    b1 = *(const v16bf*)(w1 + kt);
    acc0 = __builtin_amdgcn_wmma_f32_16x16x32_bf16(false, a, false, c0,
                                                   (short)0, acc0, false, false);
    acc1 = __builtin_amdgcn_wmma_f32_16x16x32_bf16(false, a, false, c1,
                                                   (short)0, acc1, false, false);
  }
  v16bf a = __builtin_shufflevector(a0, a1, 0,1,2,3,4,5,6,7,8,9,10,11,12,13,14,15);
  acc0 = __builtin_amdgcn_wmma_f32_16x16x32_bf16(false, a, false, b0,
                                                 (short)0, acc0, false, false);
  acc1 = __builtin_amdgcn_wmma_f32_16x16x32_bf16(false, a, false, b1,
                                                 (short)0, acc1, false, false);

  float* crow = C + (size_t)kh * 32 * ldc
                  + (size_t)(m_tile * 16 + g * 8) * ldc + n0 * 16 + r;
#pragma unroll
  for (int v = 0; v < 8; ++v) crow[(size_t)v * ldc] = acc0[v];
#pragma unroll
  for (int v = 0; v < 8; ++v) crow[(size_t)v * ldc + 16] = acc1[v];
}

// ---------------- attention ----------------

__global__ void __launch_bounds__(256)
scores_kernel(const float* __restrict__ enc, const float* __restrict__ h,
              float* __restrict__ scores) {
  int gw = blockIdx.x * 8 + (threadIdx.x >> 5);   // one wave per (b, t_enc)
  int lane = threadIdx.x & 31;
  int b = gw >> 9, te = gw & 511;
  const float* e  = enc + (size_t)(b * TENC + te) * HID;
  const float* hb = h + (size_t)b * HID;
  float acc = 0.f;
  for (int k = lane; k < HID; k += 32) acc += e[k] * hb[k];
#pragma unroll
  for (int off = 16; off > 0; off >>= 1) acc += __shfl_xor(acc, off, 32);
  if (lane == 0) scores[b * TENC + te] = acc;
}

__global__ void __launch_bounds__(256)
softmax_kernel(const float* __restrict__ scores, const int* __restrict__ lens,
               float* __restrict__ alpha, float* __restrict__ out_alphas, int t) {
  int b = blockIdx.x, tid = threadIdx.x;
  __shared__ float red[256];
  int len = lens[b];
  float s0 = (tid       < len) ? scores[b * TENC + tid]       : -1e30f;
  float s1 = (tid + 256 < len) ? scores[b * TENC + tid + 256] : -1e30f;
  red[tid] = fmaxf(s0, s1); __syncthreads();
  for (int s = 128; s > 0; s >>= 1) {
    if (tid < s) red[tid] = fmaxf(red[tid], red[tid + s]);
    __syncthreads();
  }
  float m = red[0]; __syncthreads();
  float p0 = (tid       < len) ? __expf(s0 - m) : 0.f;
  float p1 = (tid + 256 < len) ? __expf(s1 - m) : 0.f;
  red[tid] = p0 + p1; __syncthreads();
  for (int s = 128; s > 0; s >>= 1) {
    if (tid < s) red[tid] += red[tid + s];
    __syncthreads();
  }
  float inv = 1.f / red[0];
  float a0 = p0 * inv, a1 = p1 * inv;
  alpha[b * TENC + tid] = a0; alpha[b * TENC + tid + 256] = a1;
  float* oa = out_alphas + (size_t)(b * TDEC + t) * TENC;
  oa[tid] = a0; oa[tid + 256] = a1;
}

__global__ void __launch_bounds__(256)
ctx_kernel(const float* __restrict__ alpha, const float* __restrict__ enc,
           float* __restrict__ ctx) {
  int b  = blockIdx.x >> 2;
  int hh = (blockIdx.x & 3) * 256 + threadIdx.x;
  __shared__ float al[TENC];
  al[threadIdx.x]       = alpha[b * TENC + threadIdx.x];
  al[threadIdx.x + 256] = alpha[b * TENC + threadIdx.x + 256];
  __syncthreads();
  const float* e = enc + (size_t)b * TENC * HID + hh;
  float acc = 0.f;
  for (int t = 0; t < TENC; ++t) acc += al[t] * e[(size_t)t * HID];
  ctx[b * HID + hh] = acc;
}

// ---------------- pack x = [emb[tok], ctx, h] in bf16 ----------------

__global__ void __launch_bounds__(256)
pack_x_kernel(const float* __restrict__ emb, const int* __restrict__ seq,
              const float* __restrict__ ctx, const float* __restrict__ h,
              __bf16* __restrict__ x, int t) {
  int i = blockIdx.x * 256 + threadIdx.x;
  if (i >= BATCH * KCOMB) return;
  int b = i / KCOMB, j = i % KCOMB;
  float v;
  if (j < EMBD) {
    int tok = seq[b * TDEC + t];
    v = emb[(size_t)tok * EMBD + j];
  } else if (j < EMBD + HID) {
    v = ctx[b * HID + (j - EMBD)];
  } else {
    v = h[b * HID + (j - EMBD - HID)];
  }
  x[i] = f2bf(v);
}

// ------- LSTM cell (sums 2 K-partials) + pack z = [h_new, ctx] in bf16 -------

__global__ void __launch_bounds__(256)
lstm_kernel(const float* __restrict__ gates, const float* __restrict__ b_ih,
            const float* __restrict__ b_hh, const float* __restrict__ ctx,
            float* __restrict__ h, float* __restrict__ c, __bf16* __restrict__ z) {
  int i = blockIdx.x * 256 + threadIdx.x;
  if (i >= BATCH * HID) return;
  int b = i / HID, hh = i % HID;
  const float* g0 = gates + (size_t)b * NGATE;
  const float* g1 = g0 + (size_t)BATCH * NGATE;   // second K-half partial
  float gi = g0[hh]           + g1[hh]           + b_ih[hh]           + b_hh[hh];
  float gf = g0[hh + HID]     + g1[hh + HID]     + b_ih[hh + HID]     + b_hh[hh + HID];
  float gg = g0[hh + 2 * HID] + g1[hh + 2 * HID] + b_ih[hh + 2 * HID] + b_hh[hh + 2 * HID];
  float go = g0[hh + 3 * HID] + g1[hh + 3 * HID] + b_ih[hh + 3 * HID] + b_hh[hh + 3 * HID];
  float si = 1.f / (1.f + __expf(-gi));
  float sf = 1.f / (1.f + __expf(-gf));
  float so = 1.f / (1.f + __expf(-go));
  float cn = sf * c[i] + si * tanhf(gg);
  float hn = so * tanhf(cn);
  c[i] = cn; h[i] = hn;
  z[(size_t)b * KCLS + hh]       = f2bf(hn);
  z[(size_t)b * KCLS + HID + hh] = f2bf(ctx[i]);
}

// -------- label-smoothed CE per token (sums 2 K-partials of logits) --------

__global__ void __launch_bounds__(256)
loss_kernel(const float* __restrict__ logits, const float* __restrict__ b_cls,
            const int* __restrict__ labels, int t,
            float* __restrict__ loss_tok, float* __restrict__ valid_tok) {
  int b = blockIdx.x, tid = threadIdx.x;
  __shared__ float red[256];
  __shared__ float sh_ly;
  if (tid == 0) sh_ly = 0.f;
  const float* l0 = logits + (size_t)b * VPAD;
  const float* l1 = l0 + (size_t)BATCH * VPAD;    // second K-half partial
  int label = labels[b * TDEC + t];
  float m = -1e30f;
  for (int j = tid; j < VOC; j += 256) m = fmaxf(m, l0[j] + l1[j] + b_cls[j]);
  red[tid] = m; __syncthreads();
  for (int s = 128; s > 0; s >>= 1) {
    if (tid < s) red[tid] = fmaxf(red[tid], red[tid + s]);
    __syncthreads();
  }
  m = red[0]; __syncthreads();
  float se = 0.f, sl = 0.f;
  for (int j = tid; j < VOC; j += 256) {
    float l = l0[j] + l1[j] + b_cls[j];
    se += __expf(l - m);
    sl += l;
    if (j == label) sh_ly = l;
  }
  red[tid] = se; __syncthreads();
  for (int s = 128; s > 0; s >>= 1) {
    if (tid < s) red[tid] += red[tid + s];
    __syncthreads();
  }
  float seT = red[0]; __syncthreads();
  red[tid] = sl; __syncthreads();
  for (int s = 128; s > 0; s >>= 1) {
    if (tid < s) red[tid] += red[tid + s];
    __syncthreads();
  }
  float slT = red[0];
  if (tid == 0) {
    float lse = m + __logf(seT);
    bool valid = label >= 0;
    float nll = lse - sh_ly;
    float smooth = lse - slT / (float)VOC;
    float per = 0.9f * nll + 0.1f * smooth;       // EPS = 0.1
    loss_tok[t * BATCH + b]  = valid ? per : 0.f;
    valid_tok[t * BATCH + b] = valid ? 1.f : 0.f;
  }
}

__global__ void __launch_bounds__(256)
finalize_kernel(const float* __restrict__ loss_tok, const float* __restrict__ valid_tok,
                float* __restrict__ out) {
  __shared__ float r0[256], r1[256];
  int tid = threadIdx.x;
  float s = 0.f, cnt = 0.f;
  for (int i = tid; i < BATCH * TDEC; i += 256) { s += loss_tok[i]; cnt += valid_tok[i]; }
  r0[tid] = s; r1[tid] = cnt; __syncthreads();
  for (int k = 128; k > 0; k >>= 1) {
    if (tid < k) { r0[tid] += r0[tid + k]; r1[tid] += r1[tid + k]; }
    __syncthreads();
  }
  if (tid == 0) out[0] = r0[0] / fmaxf(r1[0], 1.f);
}

// ---------------- host driver ----------------

extern "C" void kernel_launch(void* const* d_in, const int* in_sizes, int n_in,
                              void* d_out, int out_size, void* d_ws, size_t ws_size,
                              hipStream_t stream) {
  const float* enc    = (const float*)d_in[0];
  const int*   seq    = (const int*)d_in[1];
  const int*   labels = (const int*)d_in[2];
  const int*   lens   = (const int*)d_in[3];
  const float* emb    = (const float*)d_in[6];
  const float* W_ih   = (const float*)d_in[7];
  const float* W_hh   = (const float*)d_in[8];
  const float* b_ih   = (const float*)d_in[9];
  const float* b_hh   = (const float*)d_in[10];
  const float* W_cls  = (const float*)d_in[11];
  const float* b_cls  = (const float*)d_in[12];
  float* out = (float*)d_out;
  float* out_alphas = out + 1;                    // [B, T_DEC, T_ENC]

  char* ws = (char*)d_ws;
  size_t off = 0;
  auto alloc = [&](size_t bytes) -> void* {
    void* p = ws + off;
    off += (bytes + 255) & ~(size_t)255;
    return p;
  };
  __bf16* Wcomb   = (__bf16*)alloc((size_t)NGATE * KCOMB * 2);
  __bf16* Wclsb   = (__bf16*)alloc((size_t)VPAD * KCLS * 2);
  float*  h       = (float*)alloc((size_t)BATCH * HID * 4);
  float*  c       = (float*)alloc((size_t)BATCH * HID * 4);
  float*  ctx     = (float*)alloc((size_t)BATCH * HID * 4);
  float*  scores  = (float*)alloc((size_t)BATCH * TENC * 4);
  float*  alpha   = (float*)alloc((size_t)BATCH * TENC * 4);
  __bf16* x       = (__bf16*)alloc((size_t)BATCH * KCOMB * 2);
  __bf16* z       = (__bf16*)alloc((size_t)BATCH * KCLS * 2);
  float*  gates   = (float*)alloc((size_t)2 * BATCH * NGATE * 4);  // 2 K-partials
  float*  logits  = (float*)alloc((size_t)2 * BATCH * VPAD * 4);   // 2 K-partials
  float*  losstk  = (float*)alloc((size_t)BATCH * TDEC * 4);
  float*  validtk = (float*)alloc((size_t)BATCH * TDEC * 4);

  // one-time (per launch) setup: bf16 weights + zero state
  conv_wcomb_kernel<<<(NGATE * KCOMB + 255) / 256, 256, 0, stream>>>(W_ih, W_hh, Wcomb);
  conv_wcls_kernel<<<(VPAD * KCLS + 255) / 256, 256, 0, stream>>>(W_cls, Wclsb);
  zero_kernel<<<(BATCH * HID + 255) / 256, 256, 0, stream>>>(h, BATCH * HID);
  zero_kernel<<<(BATCH * HID + 255) / 256, 256, 0, stream>>>(c, BATCH * HID);
  zero_kernel<<<(BATCH * TDEC + 255) / 256, 256, 0, stream>>>(losstk, BATCH * TDEC);
  zero_kernel<<<(BATCH * TDEC + 255) / 256, 256, 0, stream>>>(validtk, BATCH * TDEC);

  const dim3 gates_grid((NGATE / 16) / 4, 2);     // 64 x 2 blocks, 512 waves
  const dim3 logits_grid((VPAD / 16) / 4, 2);     // 80 x 2 blocks, 640 waves

  for (int t = 0; t < TDEC; ++t) {
    scores_kernel<<<BATCH * TENC / 8, 256, 0, stream>>>(enc, h, scores);
    softmax_kernel<<<BATCH, 256, 0, stream>>>(scores, lens, alpha, out_alphas, t);
    ctx_kernel<<<BATCH * 4, 256, 0, stream>>>(alpha, enc, ctx);
    pack_x_kernel<<<(BATCH * KCOMB + 255) / 256, 256, 0, stream>>>(emb, seq, ctx, h, x, t);
    wmma_gemm_bf16<<<gates_grid, 128, 0, stream>>>(x, Wcomb, gates, KCOMB, NGATE);
    lstm_kernel<<<(BATCH * HID + 255) / 256, 256, 0, stream>>>(gates, b_ih, b_hh,
                                                               ctx, h, c, z);
    wmma_gemm_bf16<<<logits_grid, 128, 0, stream>>>(z, Wclsb, logits, KCLS, VPAD);
    loss_kernel<<<BATCH, 256, 0, stream>>>(logits, b_cls, labels, t, losstk, validtk);
  }
  finalize_kernel<<<1, 256, 0, stream>>>(losstk, validtk, out);
}
